// HGTExplicit_14894946582735
// MI455X (gfx1250) — matrix operand order repaired and checked by reading, not compiled, not run on hardware
//
#include <hip/hip_runtime.h>
#include <math.h>

#define HID 128
#define HEADS 16
#define HD 8

typedef __attribute__((ext_vector_type(16))) __bf16 v16bf;
typedef __attribute__((ext_vector_type(8)))  float  v8f;
typedef __attribute__((ext_vector_type(4)))  unsigned v4u;

enum { PRE_GELU = 1, EPI_LEAKY = 2, EPI_SKIP = 4 };

__device__ __forceinline__ float gelu_exact(float x) {
  return 0.5f * x * (1.0f + erff(x * 0.7071067811865475f));
}

__device__ __forceinline__ unsigned pk_bf16(float a, float b) {
  union { unsigned u; __bf16 h[2]; } p;
  p.h[0] = (__bf16)a;
  p.h[1] = (__bf16)b;
  return p.u;
}

__device__ __forceinline__ void atomicMaxFloat(float* addr, float val) {
  if (val >= 0.0f) atomicMax((int*)addr, __float_as_int(val));
  else             atomicMin((unsigned int*)addr, (unsigned int)__float_as_int(val));
}

// ---------------------------------------------------------------------------
// Operand packing. A: [M,K] fp32 -> [M,K/2] packed bf16-pair dwords (pairs
// along K are contiguous -> coalesced float2 loads / dword stores). Optional
// exact-gelu folded here (runs once per element, not once per GEMM block).
// ---------------------------------------------------------------------------
__global__ void hgt_packA(const float* __restrict__ A, unsigned* __restrict__ Ap,
                          size_t npairs, int gelu)
{
  size_t i = (size_t)blockIdx.x * blockDim.x + threadIdx.x;
  if (i >= npairs) return;
  float2 v = *(const float2*)(A + 2 * i);
  float a = v.x, b = v.y;
  if (gelu) { a = gelu_exact(a); b = gelu_exact(b); }
  Ap[i] = pk_bf16(a, b);
}

// W: [K,N] fp32 -> Wp [N][K/2] packed pairs (n-major), matching the WMMA
// B-fragment per-lane layout so fragments load as two b128s per lane.
__global__ void hgt_packW(const float* __restrict__ W, unsigned* __restrict__ Wp,
                          int Kp, int N)
{
  int i = blockIdx.x * blockDim.x + threadIdx.x;
  if (i >= N * Kp) return;
  int n = i / Kp, kp = i - n * Kp;
  Wp[i] = pk_bf16(W[(size_t)(2 * kp) * N + n], W[(size_t)(2 * kp + 1) * N + n]);
}

// ---------------------------------------------------------------------------
// FAST bf16 WMMA GEMM on pre-packed operands. Requires K%32==0, N%16==0.
// Block = 256 threads (8 waves), tile = 32 rows x 128 cols; each wave owns a
// 16-col strip with TWO accumulators sharing one B fragment.
// No LDS, no barriers: fragments are direct global_load_b128 from the packed
// layouts; the K-loop is pure vmem + wmma and fully pipelineable.
// ---------------------------------------------------------------------------
__global__ void __launch_bounds__(256)
hgt_wmma_gemm_fast(const unsigned* __restrict__ Ap, const unsigned* __restrict__ Wp,
                   const float* __restrict__ bias, const float* __restrict__ skip_x,
                   const float* __restrict__ skip_s, float* __restrict__ C,
                   int M, int Kp /* = K/2 pairs */, int N, int flags)
{
  const int lane = threadIdx.x & 31;
  const int wave = threadIdx.x >> 5;
  const int m0   = blockIdx.x * 32;
  const int nb0  = blockIdx.y * 128;
  const int half = lane >> 4;
  const int l16  = lane & 15;

  const int n  = nb0 + wave * 16 + l16;
  int nc = n;            if (nc > N - 1) nc = N - 1;       // clamp, stores guarded
  int r0 = m0 + l16;     if (r0 > M - 1) r0 = M - 1;       // tail clamp
  int r1 = m0 + 16 + l16; if (r1 > M - 1) r1 = M - 1;

  const unsigned* a0p = Ap + (size_t)r0 * Kp;
  const unsigned* a1p = Ap + (size_t)r1 * Kp;
  const unsigned* bp  = Wp + (size_t)nc * Kp;

  v8f acc0 = {}, acc1 = {};

  for (int kp = 0; kp < Kp; kp += 16) {
    union { v16bf h; v4u q[2]; } ua0, ua1, ub;
    ua0.q[0] = *(const v4u*)(a0p + kp + half * 4);
    ua0.q[1] = *(const v4u*)(a0p + kp + 8 + half * 4);
    ua1.q[0] = *(const v4u*)(a1p + kp + half * 4);
    ua1.q[1] = *(const v4u*)(a1p + kp + 8 + half * 4);
    ub.q[0]  = *(const v4u*)(bp  + kp + half * 4);
    ub.q[1]  = *(const v4u*)(bp  + kp + 8 + half * 4);

    acc0 = __builtin_amdgcn_wmma_f32_16x16x32_bf16(false, ua0.h, false, ub.h,
                                                   (short)0, acc0, false, false);
    acc1 = __builtin_amdgcn_wmma_f32_16x16x32_bf16(false, ua1.h, false, ub.h,
                                                   (short)0, acc1, false, false);
  }

  // ---- epilogue ----
  if (n < N) {
    float sg = 0.0f;
    if (flags & EPI_SKIP) { float s = *skip_s; sg = 1.0f / (1.0f + expf(-s)); }
    const float bv = bias ? bias[n] : 0.0f;
#pragma unroll
    for (int r = 0; r < 8; ++r) {
      int row0 = m0 + r + half * 8;
      int row1 = row0 + 16;
      float v0 = acc0[r] + bv;
      float v1 = acc1[r] + bv;
      if (flags & EPI_SKIP) {
        if (row0 < M) v0 = sg * v0 + (1.0f - sg) * skip_x[(size_t)row0 * N + n];
        if (row1 < M) v1 = sg * v1 + (1.0f - sg) * skip_x[(size_t)row1 * N + n];
      }
      if (flags & EPI_LEAKY) {
        v0 = (v0 > 0.0f) ? v0 : 0.01f * v0;
        v1 = (v1 > 0.0f) ? v1 : 0.01f * v1;
      }
      if (row0 < M) C[(size_t)row0 * N + n] = v0;
      if (row1 < M) C[(size_t)row1 * N + n] = v1;
    }
  }
}

// ---------------------------------------------------------------------------
// Generic (guarded) WMMA GEMM for ragged shapes (conv1 projections with
// K in {25,5,2}). Same math, bounds-checked loads, LDS-staged A.
// ---------------------------------------------------------------------------
__global__ void __launch_bounds__(256)
hgt_wmma_gemm_gen(const float* __restrict__ A, const float* __restrict__ W,
                  const float* __restrict__ bias, float* __restrict__ C,
                  int M, int K, int N)
{
  __shared__ __bf16 lA[16][40];

  const int lane = threadIdx.x & 31;
  const int wave = threadIdx.x >> 5;
  const int m0   = blockIdx.x * 16;
  const int n0   = (blockIdx.y * 8 + wave) * 16;
  const int half = lane >> 4;
  const int l16  = lane & 15;

  v8f acc = {};

  for (int kb = 0; kb < K; kb += 32) {
    for (int t = threadIdx.x; t < 512; t += 256) {
      int r = t >> 5, c = t & 31;
      float v = 0.0f;
      int gr = m0 + r, gc = kb + c;
      if (gr < M && gc < K) v = A[(size_t)gr * K + gc];
      lA[r][c] = (__bf16)v;
    }
    __syncthreads();

    v16bf af;
#pragma unroll
    for (int i = 0; i < 8; ++i) {
      int kofs = (i < 4) ? (half * 8 + 2 * i) : (16 + half * 8 + 2 * (i - 4));
      af[2 * i]     = lA[l16][kofs];
      af[2 * i + 1] = lA[l16][kofs + 1];
    }

    v16bf bf;
    int n = n0 + l16;
#pragma unroll
    for (int j = 0; j < 8; ++j) {
      int k0 = kb + ((j < 4) ? (half * 8 + 2 * j) : (16 + half * 8 + 2 * (j - 4)));
      float b0 = 0.0f, b1 = 0.0f;
      if (n < N) {
        if (k0 < K)     b0 = W[(size_t)k0 * N + n];
        if (k0 + 1 < K) b1 = W[(size_t)(k0 + 1) * N + n];
      }
      bf[2 * j]     = (__bf16)b0;
      bf[2 * j + 1] = (__bf16)b1;
    }

    acc = __builtin_amdgcn_wmma_f32_16x16x32_bf16(false, af, false, bf,
                                                  (short)0, acc, false, false);
    __syncthreads();
  }

  int n = n0 + l16;
#pragma unroll
  for (int r = 0; r < 8; ++r) {
    int row = m0 + r + half * 8;
    if (row < M && n < N) {
      float v = acc[r];
      if (bias) v += bias[n];
      C[(size_t)row * N + n] = v;
    }
  }
}

// ---------------------------------------------------------------------------
// Per-head relation transform: out[n,h,f] = sum_d in[n,h,d] * rel[h,d,f]
// ---------------------------------------------------------------------------
__global__ void hgt_rel(const float* __restrict__ in, const float* __restrict__ rel,
                        float* __restrict__ out, int Nn)
{
  int idx = blockIdx.x * blockDim.x + threadIdx.x;
  if (idx >= Nn * HID) return;
  int nf = idx & (HID - 1);
  int n  = idx >> 7;
  int h  = nf >> 3;
  int f  = nf & 7;
  const float* s = in + (size_t)n * HID + h * HD;
  const float* a = rel + h * HD * HD + f;
  float acc = 0.0f;
#pragma unroll
  for (int d = 0; d < HD; ++d) acc += s[d] * a[d * HD];
  out[idx] = acc;
}

__global__ void hgt_fill(float* __restrict__ p, float v, size_t n)
{
  size_t i = (size_t)blockIdx.x * blockDim.x + threadIdx.x;
  if (i < n) p[i] = v;
}

__global__ void hgt_alpha(const int* __restrict__ ei, int E,
                          const float* __restrict__ q, const float* __restrict__ krel,
                          const float* __restrict__ prel, float scale,
                          float* __restrict__ alpha, float* __restrict__ amax)
{
  int idx = blockIdx.x * blockDim.x + threadIdx.x;
  if (idx >= E * HEADS) return;
  int e = idx >> 4, h = idx & 15;
  int si = ei[e], di = ei[E + e];
  const float* qp = q    + (size_t)di * HID + h * HD;
  const float* kp = krel + (size_t)si * HID + h * HD;
  float a = 0.0f;
#pragma unroll
  for (int d = 0; d < HD; ++d) a += qp[d] * kp[d];
  a *= prel[h] * scale;
  alpha[idx] = a;
  atomicMaxFloat(&amax[di * HEADS + h], a);
}

__global__ void hgt_expsum(const int* __restrict__ ei, int E,
                           const float* __restrict__ amax, float* __restrict__ alpha,
                           float* __restrict__ den)
{
  int idx = blockIdx.x * blockDim.x + threadIdx.x;
  if (idx >= E * HEADS) return;
  int e = idx >> 4, h = idx & 15;
  int di = ei[E + e];
  float ex = expf(alpha[idx] - amax[di * HEADS + h]);
  alpha[idx] = ex;
  atomicAdd(&den[di * HEADS + h], ex);
}

__global__ void hgt_scatter(const int* __restrict__ ei, int E,
                            const float* __restrict__ alpha, const float* __restrict__ den,
                            const float* __restrict__ vrel, float* __restrict__ agg)
{
  int idx = blockIdx.x * blockDim.x + threadIdx.x;
  if (idx >= E * HEADS) return;
  int e = idx >> 4, h = idx & 15;
  int si = ei[e], di = ei[E + e];
  float w = alpha[idx] / (den[di * HEADS + h] + 1e-16f);
  const float* vp = vrel + (size_t)si * HID + h * HD;
  float* ap = agg + (size_t)di * HID + h * HD;
#pragma unroll
  for (int d = 0; d < HD; ++d) atomicAdd(&ap[d], vp[d] * w);
}

__global__ void hgt_out2(const float* __restrict__ h, const float* __restrict__ w2,
                         const float* __restrict__ b2, float* __restrict__ out, int nOut)
{
  int i = blockIdx.x * blockDim.x + threadIdx.x;
  if (i >= nOut) return;
  const float* r = h + (size_t)i * 12 * 64;
  float acc = b2[0];
#pragma unroll
  for (int j = 0; j < 64; ++j) acc += r[j] * w2[j];
  out[i] = acc;
}

// ---------------------------------------------------------------------------
// Host orchestration. Input layout: 0-4 x_{stock,other,connect,financing,
// selling}; 5-12 edge arrays ([2,E] int32, ET1 order); 13.. params pytree
// leaves (dict keys sorted alphabetically at every level).
// ---------------------------------------------------------------------------
extern "C" void kernel_launch(void* const* d_in, const int* in_sizes, int n_in,
                              void* d_out, int out_size, void* d_ws, size_t ws_size,
                              hipStream_t stream)
{
  (void)n_in; (void)out_size; (void)ws_size;
  auto F  = [&](int i) { return (const float*)d_in[i]; };
  auto Ip = [&](int i) { return (const int*)d_in[i]; };

  const int NN[5]   = {50000, 20000, 5000, 5000, 5000};
  const int INCH[5] = {25, 768, 5, 2, 2};

  float* ws = (float*)d_ws;
  size_t off = 0;
  auto alloc = [&](size_t n) { float* p = ws + off; off += n; return p; };

  float* cur[2]; float* nxt[2]; float* agg[2];
  cur[0] = alloc((size_t)NN[0] * HID); cur[1] = alloc((size_t)NN[1] * HID);
  nxt[0] = alloc((size_t)NN[0] * HID); nxt[1] = alloc((size_t)NN[1] * HID);
  float *kf[5], *qf[5], *vf[5];
  for (int t = 0; t < 5; ++t) {
    kf[t] = alloc((size_t)NN[t] * HID);
    qf[t] = alloc((size_t)NN[t] * HID);
    vf[t] = alloc((size_t)NN[t] * HID);
  }
  float* krel  = alloc((size_t)NN[0] * HID);
  float* vrel  = alloc((size_t)NN[0] * HID);
  float* albuf = alloc((size_t)200000 * HEADS);
  float* amax  = alloc((size_t)NN[0] * HEADS);
  float* den   = alloc((size_t)NN[0] * HEADS);
  agg[0] = alloc((size_t)NN[0] * HID); agg[1] = alloc((size_t)NN[1] * HID);
  float* out1  = alloc((size_t)NN[0] * 64);
  unsigned* ApU = (unsigned*)alloc((size_t)20000 * 384);  // max M*K/2 (x_other)
  unsigned* WpU = (unsigned*)alloc((size_t)128 * 384);    // max N*K/2

  auto packA = [&](const float* A, int M, int K, int gelu) {
    size_t np = (size_t)M * (K / 2);
    hgt_packA<<<(unsigned)((np + 255) / 256), 256, 0, stream>>>(A, ApU, np, gelu);
  };
  auto fastg = [&](const float* Wf, const float* b, const float* sx, const float* ss,
                   float* C, int M, int K, int N, int flags) {
    int Kp = K / 2;
    int npw = N * Kp;
    hgt_packW<<<(npw + 255) / 256, 256, 0, stream>>>(Wf, WpU, Kp, N);
    dim3 grid((M + 31) / 32, (N + 127) / 128);
    hgt_wmma_gemm_fast<<<grid, dim3(256), 0, stream>>>(ApU, WpU, b, sx, ss, C,
                                                       M, Kp, N, flags);
  };
  auto gemm_gen = [&](const float* A, const float* Wf, const float* b, float* C,
                      int M, int K, int N) {
    dim3 grid((M + 15) / 16, (N + 127) / 128);
    hgt_wmma_gemm_gen<<<grid, dim3(256), 0, stream>>>(A, Wf, b, C, M, K, N);
  };
  auto fill = [&](float* p, float v, size_t n) {
    hgt_fill<<<(unsigned)((n + 255) / 256), 256, 0, stream>>>(p, v, n);
  };

  struct ETm { int e, s, d, a, m, p; };
  const float scale = 0.35355339059327373f;  // 1/sqrt(HD)

  auto do_edges = [&](const ETm* ets, int nET) {
    for (int i = 0; i < nET; ++i) {
      const ETm& et = ets[i];
      int E  = in_sizes[et.e] / 2;
      int Ns = NN[et.s], Nd = NN[et.d];
      int thr = Ns * HID;
      hgt_rel<<<(thr + 255) / 256, 256, 0, stream>>>(kf[et.s], F(et.a), krel, Ns);
      hgt_rel<<<(thr + 255) / 256, 256, 0, stream>>>(vf[et.s], F(et.m), vrel, Ns);
      fill(amax, -INFINITY, (size_t)Nd * HEADS);
      fill(den, 0.0f, (size_t)Nd * HEADS);
      int eth = E * HEADS;
      hgt_alpha<<<(eth + 255) / 256, 256, 0, stream>>>(Ip(et.e), E, qf[et.d], krel,
                                                       F(et.p), scale, albuf, amax);
      hgt_expsum<<<(eth + 255) / 256, 256, 0, stream>>>(Ip(et.e), E, amax, albuf, den);
      hgt_scatter<<<(eth + 255) / 256, 256, 0, stream>>>(Ip(et.e), E, albuf, den, vrel,
                                                         agg[et.d]);
    }
  };

  // ---------------- conv1 ----------------
  const int c1kb[5] = {66, 51, 37, 43, 58}, c1kw[5] = {67, 52, 38, 44, 59};
  const int c1qb[5] = {68, 53, 39, 45, 60}, c1qw[5] = {69, 54, 40, 46, 61};
  const int c1vb[5] = {71, 56, 41, 47, 62}, c1vw[5] = {72, 57, 42, 48, 63};
  // 'other' (t=1): K=768 -> packed fast path, A packed once for k/q/v
  packA(F(1), NN[1], INCH[1], 0);
  fastg(F(c1kw[1]), F(c1kb[1]), nullptr, nullptr, kf[1], NN[1], INCH[1], HID, 0);
  fastg(F(c1qw[1]), F(c1qb[1]), nullptr, nullptr, qf[1], NN[1], INCH[1], HID, 0);
  fastg(F(c1vw[1]), F(c1vb[1]), nullptr, nullptr, vf[1], NN[1], INCH[1], HID, 0);
  // ragged-K types via guarded kernel
  for (int t = 0; t < 5; ++t) {
    if (t == 1) continue;
    const float* x = F(t);
    gemm_gen(x, F(c1kw[t]), F(c1kb[t]), kf[t], NN[t], INCH[t], HID);
    gemm_gen(x, F(c1qw[t]), F(c1qb[t]), qf[t], NN[t], INCH[t], HID);
    gemm_gen(x, F(c1vw[t]), F(c1vb[t]), vf[t], NN[t], INCH[t], HID);
  }
  fill(agg[0], 0.0f, (size_t)NN[0] * HID);
  fill(agg[1], 0.0f, (size_t)NN[1] * HID);
  const ETm e1[8] = {{5, 0, 0, 34, 35, 36}, {6, 2, 0, 13, 14, 15},
                     {7, 3, 0, 16, 17, 18}, {8, 4, 0, 25, 26, 27},
                     {9, 0, 0, 31, 32, 33}, {10, 0, 1, 28, 29, 30},
                     {11, 1, 0, 22, 23, 24}, {12, 1, 1, 19, 20, 21}};
  do_edges(e1, 8);
  // a_lin: gelu folded into packA; no skip (in_ch != 128); leaky epilogue
  packA(agg[0], NN[0], HID, 1);
  fastg(F(65), F(64), nullptr, nullptr, cur[0], NN[0], HID, HID, EPI_LEAKY);
  packA(agg[1], NN[1], HID, 1);
  fastg(F(50), F(49), nullptr, nullptr, cur[1], NN[1], HID, HID, EPI_LEAKY);

  // ---------------- conv2 / conv3 ----------------
  const ETm e2[5] = {{5, 0, 0, 85, 86, 87}, {9, 0, 0, 82, 83, 84},
                     {10, 0, 1, 79, 80, 81}, {11, 1, 0, 76, 77, 78},
                     {12, 1, 1, 73, 74, 75}};
  const ETm e3[5] = {{5, 0, 0, 118, 119, 120}, {9, 0, 0, 115, 116, 117},
                     {10, 0, 1, 112, 113, 114}, {11, 1, 0, 109, 110, 111},
                     {12, 1, 1, 106, 107, 108}};
  struct NTp { int kb, kw, qb, qw, vb, vw, ab, aw, skip; };
  const NTp n2[2] = {{99, 100, 101, 102, 104, 105, 97, 98, 103},
                     {90, 91, 92, 93, 95, 96, 88, 89, 94}};
  const NTp n3[2] = {{132, 133, 134, 135, 137, 138, 130, 131, 136},
                     {123, 124, 125, 126, 128, 129, 121, 122, 127}};

  for (int layer = 0; layer < 2; ++layer) {
    const ETm* ee = layer ? e3 : e2;
    const NTp* np = layer ? n3 : n2;
    for (int t = 0; t < 2; ++t) {
      packA(cur[t], NN[t], HID, 0);   // shared by k/q/v
      fastg(F(np[t].kw), F(np[t].kb), nullptr, nullptr, kf[t], NN[t], HID, HID, 0);
      fastg(F(np[t].qw), F(np[t].qb), nullptr, nullptr, qf[t], NN[t], HID, HID, 0);
      fastg(F(np[t].vw), F(np[t].vb), nullptr, nullptr, vf[t], NN[t], HID, HID, 0);
    }
    fill(agg[0], 0.0f, (size_t)NN[0] * HID);
    fill(agg[1], 0.0f, (size_t)NN[1] * HID);
    do_edges(ee, 5);
    for (int t = 0; t < 2; ++t) {
      packA(agg[t], NN[t], HID, 1);   // gelu folded
      fastg(F(np[t].aw), F(np[t].ab), cur[t], F(np[t].skip), nxt[t],
            NN[t], HID, HID, EPI_SKIP | EPI_LEAKY);
    }
    float* t0 = cur[0]; cur[0] = nxt[0]; nxt[0] = t0;
    float* t1 = cur[1]; cur[1] = nxt[1]; nxt[1] = t1;
  }

  // ---------------- head ----------------
  packA(cur[0], NN[0], HID, 0);
  fastg(F(140), F(139), nullptr, nullptr, out1, NN[0], HID, 64, EPI_LEAKY);
  int nOut = (NN[0] + 11) / 12;
  hgt_out2<<<(nOut + 255) / 256, 256, 0, stream>>>(out1, F(142), F(141),
                                                   (float*)d_out, nOut);
}
